// MagiAttention_49263274885542
// MI455X (gfx1250) — matrix operand order repaired and checked by reading, not compiled, not run on hardware
//
#include <hip/hip_runtime.h>
#include <hip/hip_bf16.h>

// MagiAttention pipeline for gfx1250 (MI455X), wave32 + WMMA + TDM/async-LDS.
//  1) rmsnorm_pre   : RMSNorm(hidden)*(w_pre+1) -> f16 activations
//  2) cast_f32_f16  : fp32 weights -> f16 (the HBM-bound stage)
//  3) gemm_f16_wmma : C = A @ W^T via v_wmma_f32_16x16x32_f16;
//                     B tile staged by TENSOR_LOAD_TO_LDS (TDM descriptor),
//                     A tile staged by GLOBAL_LOAD_ASYNC_TO_LDS_B128
//  4) qkv_post      : per-head RMSNorm + RoPE + inverse-permute, pack f16 Q/K/V
//  5) flash_attn    : online-softmax flash attention, WMMA for QK^T and PV,
//                     async-LDS staging for the K tile
//  6) gemm_f16_wmma : output projection (per-expert) -> fp32 d_out
// Workspace use: ~190 MB.

#define N_TOK 2048
#define HID   3072
#define HQ    24
#define HKV   8
#define HD    128
#define QKV_OUT 5144   /* HQ*HD + 2*HKV*HD + HQ */
#define S0    1024
#define EPS_F 1e-6f

typedef _Float16 f16;
typedef __attribute__((ext_vector_type(16))) _Float16 v16h;
typedef __attribute__((ext_vector_type(8)))  _Float16 v8h;
typedef __attribute__((ext_vector_type(8)))  float    v8f;

// ---- CDNA5 data-mover feature detection ---------------------------------

#if defined(__has_builtin)
#if __has_builtin(__builtin_amdgcn_global_load_async_to_lds_b128) && \
    __has_builtin(__builtin_amdgcn_s_wait_asynccnt)
#define HAVE_ASYNC_LDS 1
#endif
#if __has_builtin(__builtin_amdgcn_tensor_load_to_lds) && \
    __has_builtin(__builtin_amdgcn_s_wait_tensorcnt)
#define HAVE_TDM 1
#endif
#endif
#ifndef HAVE_ASYNC_LDS
#define HAVE_ASYNC_LDS 0
#endif
#ifndef HAVE_TDM
#define HAVE_TDM 0
#endif

typedef __attribute__((address_space(3))) f16 las_f16;

#if HAVE_ASYNC_LDS
// Builtin signature (per hipcc diagnostic): param0 is a pointer to a GCC-style
// int4 vector in the global (__device__) address space; param1 is the LDS-side
// pointer.  Cast through exactly those pointee types.
typedef int ivec4 __attribute__((vector_size(16)));
typedef __attribute__((address_space(1))) ivec4 gas_ivec4;
typedef __attribute__((address_space(3))) ivec4 las_ivec4;
static __device__ __forceinline__ void async_cp16(const void* g, void* l) {
  __builtin_amdgcn_global_load_async_to_lds_b128(
      (gas_ivec4*)g, (las_ivec4*)l, 0, 0);
}
#endif

#if HAVE_TDM
typedef unsigned int u32x4 __attribute__((ext_vector_type(4)));
typedef int          i32x4 __attribute__((ext_vector_type(4)));
typedef int          i32x8 __attribute__((ext_vector_type(8)));

// Issue a 2-D TDM load: tile_d0 (contiguous f16 elems) x tile_d1 (rows),
// row stride stride_d0 elems; tensor dims give hardware OOB clamping.
// D# layout per CDNA5 ISA 8.3/8.4 (group0: flags/lds/global addr, group1:
// data_size + dims + strides).  One wave issues; EXEC is ignored by TDM.
static __device__ __forceinline__ void tdm_load_2d(
    const void* gaddr, unsigned lds_byte_off,
    unsigned tile_d0, unsigned tile_d1,
    unsigned tensor_d0, unsigned tensor_d1, unsigned stride_d0) {
  unsigned long long ga = (unsigned long long)gaddr;
  u32x4 g0;
  g0[0] = 1u;                                        // count=1 (valid user D#)
  g0[1] = lds_byte_off;                              // lds_addr
  g0[2] = (unsigned)(ga & 0xffffffffu);              // global_addr[31:0]
  g0[3] = (unsigned)((ga >> 32) & 0x1ffffffu) | (2u << 30);  // [56:32] | type=2
  i32x8 g1;
  g1[0] = (int)(1u << 16);                           // data_size=2B; mask=0
  g1[1] = (int)((tensor_d0 & 0xffffu) << 16);        // tensor_dim0[15:0]
  g1[2] = (int)(((tensor_d0 >> 16) & 0xffffu) | ((tensor_d1 & 0xffffu) << 16));
  g1[3] = (int)(((tensor_d1 >> 16) & 0xffffu) | ((tile_d0 & 0xffffu) << 16));
  g1[4] = (int)(tile_d1 & 0xffffu);                  // tile_dim1; tile_dim2=0
  g1[5] = (int)stride_d0;                            // dim0_stride[31:0]
  g1[6] = 0;                                         // dim0_stride[47:32], dim1_stride lo
  g1[7] = 0;
  i32x4 z4 = {};
#if defined(__clang_major__) && __clang_major__ >= 23
  i32x8 z8 = {};
  __builtin_amdgcn_tensor_load_to_lds(g0, g1, z4, z4, z8, 0);
#else
  __builtin_amdgcn_tensor_load_to_lds(g0, g1, z4, z4, 0);
#endif
}
#endif

// ---- helpers -------------------------------------------------------------

// A-operand fragment (16x32 f16): lane<16 holds row l, K {0..7,16..23};
// lane>=16 holds row l-16, K {8..15,24..31}.
static __device__ __forceinline__ v16h ld_afrag(const f16* lo, const f16* hi) {
  v8h a = *(const v8h*)lo;
  v8h b = *(const v8h*)hi;
  v16h r;
#pragma unroll
  for (int i = 0; i < 8; ++i) { r[i] = a[i]; r[i + 8] = b[i]; }
  return r;
}

static __device__ __forceinline__ float allred_sum32(float v) {
#pragma unroll
  for (int m = 1; m <= 16; m <<= 1) v += __shfl_xor(v, m, 32);
  return v;
}
static __device__ __forceinline__ float allred_max16(float v) {
#pragma unroll
  for (int m = 1; m <= 8; m <<= 1) v = fmaxf(v, __shfl_xor(v, m, 32));
  return v;
}
static __device__ __forceinline__ float allred_sum16(float v) {
#pragma unroll
  for (int m = 1; m <= 8; m <<= 1) v += __shfl_xor(v, m, 32);
  return v;
}

// ---- 1) pre-RMSNorm + f16 cast ------------------------------------------

__global__ void rmsnorm_pre(const float* __restrict__ x,
                            const float* __restrict__ w_pre,
                            f16* __restrict__ out) {
  const int row = blockIdx.x;
  const int expert = row >= S0;
  const float* xr = x + (size_t)row * HID;
  float ss = 0.f;
  for (int c = threadIdx.x; c < HID; c += 256) { float v = xr[c]; ss += v * v; }
  ss = allred_sum32(ss);
  __shared__ float red[8];
  if ((threadIdx.x & 31) == 0) red[threadIdx.x >> 5] = ss;
  __syncthreads();
  float tot = 0.f;
#pragma unroll
  for (int i = 0; i < 8; ++i) tot += red[i];
  const float rms = rsqrtf(tot * (1.0f / HID) + EPS_F);
  const float* w = w_pre + (size_t)expert * HID;
  for (int c = threadIdx.x; c < HID; c += 256)
    out[(size_t)row * HID + c] = (f16)(xr[c] * rms * (w[c] + 1.0f));
}

// ---- 2) weight down-conversion ------------------------------------------

__global__ void cast_f32_f16(const float* __restrict__ in, f16* __restrict__ out, int n) {
  for (int i = blockIdx.x * blockDim.x + threadIdx.x; i < n; i += gridDim.x * blockDim.x)
    out[i] = (f16)in[i];
}

// ---- 3/6) tiled WMMA GEMM: C[M,Nout] = A[M,K] @ W[Nout,K]^T --------------
// block = 256 threads = 8 waves in a 2(M) x 4(N) grid of 16x16 WMMA tiles.
// B tile (64x32 f16, strided) comes in via TDM; A tile via async-LDS b128.

#define GEMM_BM 32
#define GEMM_BN 64
#define GEMM_BK 32

__global__ void gemm_f16_wmma(const f16* __restrict__ A, const f16* __restrict__ W,
                              float* __restrict__ C, int M, int Nout, int K) {
  (void)M;
  __shared__ __attribute__((aligned(64))) f16 sA[GEMM_BM * GEMM_BK];
  __shared__ __attribute__((aligned(64))) f16 sB[GEMM_BN * GEMM_BK];
  const int tid = threadIdx.x;
  const int wave = tid >> 5, lane = tid & 31;
  const int l16 = lane & 15, lh = lane >> 4;
  const int wm = wave >> 2, wn = wave & 3;
  const int tileM = blockIdx.y * GEMM_BM;
  const int tileN = blockIdx.x * GEMM_BN;
  const int rowA = tileM + wm * 16;
  const int colB = tileN + wn * 16;

  v8f acc = {};
  const int srow = tid >> 2;            // staging row
  const int scol = (tid & 3) * 8;       // staging chunk (8 halfs = 16B)
  const int wrow = tileN + srow < Nout ? tileN + srow : 0;  // clamp edge rows
#if HAVE_TDM
  const unsigned sB_off = (unsigned)(las_f16*)&sB[0];
#endif

  for (int k0 = 0; k0 < K; k0 += GEMM_BK) {
    // ---- stage B: 64 weight rows x 32 halfs
#if HAVE_TDM
    if (wave == 0)
      tdm_load_2d(&W[(size_t)tileN * K + k0], sB_off,
                  GEMM_BK, GEMM_BN,
                  (unsigned)(K - k0), (unsigned)(Nout - tileN), (unsigned)K);
#elif HAVE_ASYNC_LDS
    async_cp16(&W[(size_t)wrow * K + k0 + scol], &sB[srow * GEMM_BK + scol]);
#else
    *(v8h*)&sB[srow * GEMM_BK + scol] =
        *(const v8h*)&W[(size_t)wrow * K + k0 + scol];
#endif
    // ---- stage A: 32 rows x 32 halfs
#if HAVE_ASYNC_LDS
    if (tid < 128)
      async_cp16(&A[(size_t)(tileM + srow) * K + k0 + scol],
                 &sA[srow * GEMM_BK + scol]);
    __builtin_amdgcn_s_wait_asynccnt(0);
#else
    if (tid < 128)
      *(v8h*)&sA[srow * GEMM_BK + scol] =
          *(const v8h*)&A[(size_t)(tileM + srow) * K + k0 + scol];
#endif
#if HAVE_TDM
    if (wave == 0) __builtin_amdgcn_s_wait_tensorcnt(0);
#endif
    if (k0 + GEMM_BK < K)
      __builtin_prefetch(&W[(size_t)wrow * K + k0 + GEMM_BK], 0, 1);
    __syncthreads();

    const int ar = wm * 16 + l16;
    const int ak = lh ? 8 : 0;
    v16h af = ld_afrag(&sA[ar * GEMM_BK + ak], &sA[ar * GEMM_BK + ak + 16]);
    const int br = wn * 16 + l16;
    const int bk = lh ? 8 : 0;
    v16h bf = ld_afrag(&sB[br * GEMM_BK + bk], &sB[br * GEMM_BK + bk + 16]);
    acc = __builtin_amdgcn_wmma_f32_16x16x32_f16(false, af, false, bf,
                                                 (short)0, acc, false, false);
    __syncthreads();
  }

  const int cc = colB + l16;
  if (cc < Nout) {
#pragma unroll
    for (int r = 0; r < 8; ++r) {
      const int rr = rowA + r + lh * 8;
      C[(size_t)rr * Nout + cc] = acc[r];
    }
  }
}

// ---- 4) q/k RMSNorm + RoPE + inverse permute, pack f16 head-major --------

__global__ void qkv_post(const float* __restrict__ qkv, const float* __restrict__ rope,
                         const float* __restrict__ wqn, const float* __restrict__ wkn,
                         const int* __restrict__ invp,
                         f16* __restrict__ qo, f16* __restrict__ ko, f16* __restrict__ vo) {
  const int t = blockIdx.x;               // time-order token
  const int tid = threadIdx.x, wave = tid >> 5, lane = tid & 31;
  const int src = invp[t];                // sorted-order row
  const int expert = src >= S0;
  const float* row = qkv + (size_t)src * QKV_OUT;
  const float* rp = rope + (size_t)t * HD;  // [0..63]=sin, [64..127]=cos

  for (int task = wave; task < HQ + HKV; task += 8) {
    const bool isq = task < HQ;
    const float* x = isq ? row + task * HD : row + HQ * HD + (task - HQ) * HD;
    const float* wn = (isq ? wqn : wkn) + expert * HD;
    float v[4], y[4];
    float ss = 0.f;
#pragma unroll
    for (int j = 0; j < 4; ++j) { const int d = lane + 32 * j; v[j] = x[d]; ss += v[j] * v[j]; }
    ss = allred_sum32(ss);
    const float rms = rsqrtf(ss * (1.0f / HD) + EPS_F);
#pragma unroll
    for (int j = 0; j < 4; ++j) { const int d = lane + 32 * j; y[j] = v[j] * rms * (wn[d] + 1.0f); }
    float o[4];
#pragma unroll
    for (int j = 0; j < 4; ++j) {        // pair (d, d+64) lives at (j, j^2)
      const int dm = lane + 32 * (j & 1);
      const float c = rp[64 + dm];
      const float s = rp[dm];
      const float rot = (j < 2) ? -y[j + 2] : y[j - 2];
      o[j] = y[j] * c + rot * s;
    }
    f16* dst = isq ? qo + ((size_t)task * N_TOK + t) * HD
                   : ko + ((size_t)(task - HQ) * N_TOK + t) * HD;
#pragma unroll
    for (int j = 0; j < 4; ++j) dst[lane + 32 * j] = (f16)o[j];
  }
  for (int idx = tid; idx < HKV * HD; idx += 256) {
    const int kh = idx >> 7, d = idx & 127;
    vo[((size_t)kh * N_TOK + t) * HD + d] =
        (f16)row[HQ * HD + HKV * HD + kh * HD + d];
  }
}

// ---- 5) flash attention (WMMA QK^T and PV, online softmax) ---------------
// grid (N/32, HKV); block 256 = 8 waves; waves 0..5 = 3 q-heads x 2 q-tiles
// sharing this block's KV tiles.

__global__ void flash_attn(const f16* __restrict__ qbf, const f16* __restrict__ kbf,
                           const f16* __restrict__ vbf, const float* __restrict__ qkv,
                           const int* __restrict__ invp, f16* __restrict__ obf) {
  __shared__ __attribute__((aligned(64))) f16 sK[32 * HD];    // 32 keys x 128
  __shared__ __attribute__((aligned(64))) f16 sVT[HD * 32];   // transposed
  __shared__ __attribute__((aligned(64))) f16 sP[6][16 * 32]; // per-wave P tile
  const int kvh = blockIdx.y;
  const int qblk = blockIdx.x * 32;
  const int tid = threadIdx.x, wave = tid >> 5, lane = tid & 31;
  const int l16 = lane & 15, lh = lane >> 4;
  const bool active = wave < 6;
  const int h = kvh * 3 + (wave % 3);
  const int qrow0 = qblk + (wave / 3) * 16;

  v16h qf[4];
  if (active) {
    const f16* qb = qbf + ((size_t)h * N_TOK + qrow0 + l16) * HD;
#pragma unroll
    for (int kc = 0; kc < 4; ++kc) {
      const int off = kc * 32 + (lh ? 8 : 0);
      qf[kc] = ld_afrag(qb + off, qb + off + 16);
    }
  }
  float mrow[8], lrow[8];
  v8f o[8] = {};
#pragma unroll
  for (int r = 0; r < 8; ++r) { mrow[r] = -3.0e38f; lrow[r] = 0.f; }
  const float scale = 0.08838834764831845f;  // 1/sqrt(128)

  for (int j0 = 0; j0 < N_TOK; j0 += 32) {
    // stage K tile (row-major): 512 x 16B chunks
#pragma unroll
    for (int it = 0; it < 2; ++it) {
      const int idx = tid + it * 256;
      const int key = idx >> 4, c = (idx & 15) * 8;
#if HAVE_ASYNC_LDS
      async_cp16(&kbf[((size_t)kvh * N_TOK + j0 + key) * HD + c],
                 &sK[key * HD + c]);
#else
      *(v8h*)&sK[key * HD + c] =
          *(const v8h*)&kbf[((size_t)kvh * N_TOK + j0 + key) * HD + c];
#endif
    }
    // stage V transposed (d-major) so PV B-frags are contiguous
#pragma unroll
    for (int it = 0; it < 16; ++it) {
      const int idx = tid + it * 256;
      const int key = idx >> 7, d = idx & 127;
      sVT[d * 32 + key] = vbf[((size_t)kvh * N_TOK + j0 + key) * HD + d];
    }
#if HAVE_ASYNC_LDS
    __builtin_amdgcn_s_wait_asynccnt(0);
#endif
    __syncthreads();

    if (active) {
      v8f s0 = {}, s1 = {};
#pragma unroll
      for (int kc = 0; kc < 4; ++kc) {
        const int koff = kc * 32 + (lh ? 8 : 0);
        v16h k0 = ld_afrag(&sK[(0 + l16) * HD + koff], &sK[(0 + l16) * HD + koff + 16]);
        v16h k1 = ld_afrag(&sK[(16 + l16) * HD + koff], &sK[(16 + l16) * HD + koff + 16]);
        s0 = __builtin_amdgcn_wmma_f32_16x16x32_f16(false, qf[kc], false, k0,
                                                    (short)0, s0, false, false);
        s1 = __builtin_amdgcn_wmma_f32_16x16x32_f16(false, qf[kc], false, k1,
                                                    (short)0, s1, false, false);
      }
#pragma unroll
      for (int r = 0; r < 8; ++r) {
        const float a = s0[r] * scale;
        const float b = s1[r] * scale;
        float mx = allred_max16(fmaxf(a, b));
        const float mnew = fmaxf(mrow[r], mx);
        const float p0 = __expf(a - mnew);
        const float p1 = __expf(b - mnew);
        const float corr = __expf(mrow[r] - mnew);
        const float rs = allred_sum16(p0 + p1);
        lrow[r] = lrow[r] * corr + rs;
        mrow[r] = mnew;
#pragma unroll
        for (int f = 0; f < 8; ++f) o[f][r] *= corr;
        const int prow = r + lh * 8;
        sP[wave][prow * 32 + l16] = (f16)p0;
        sP[wave][prow * 32 + 16 + l16] = (f16)p1;
      }
      const f16* pb = &sP[wave][l16 * 32];
      const int ak = lh ? 8 : 0;
      v16h pf = ld_afrag(pb + ak, pb + ak + 16);
#pragma unroll
      for (int f = 0; f < 8; ++f) {
        const int d0 = f * 16;
        v16h vf = ld_afrag(&sVT[(d0 + l16) * 32 + (lh ? 8 : 0)], &sVT[(d0 + l16) * 32 + (lh ? 8 : 0) + 16]);
        o[f] = __builtin_amdgcn_wmma_f32_16x16x32_f16(false, pf, false, vf,
                                                      (short)0, o[f], false, false);
      }
    }
    __syncthreads();
  }

  if (active) {
#pragma unroll
    for (int r = 0; r < 8; ++r) {
      const int t = qrow0 + r + lh * 8;     // time-order query row
      const int i = invp[t];                // sorted-order output row
      const float g = qkv[(size_t)i * QKV_OUT + HQ * HD + 2 * HKV * HD + h];
      const float gs = 1.0f / (1.0f + __expf(-g));
      const float il = 1.0f / lrow[r];
#pragma unroll
      for (int f = 0; f < 8; ++f)
        obf[(size_t)i * (HQ * HD) + h * HD + f * 16 + l16] = (f16)(o[f][r] * il * gs);
    }
  }
}

// ---- host launcher -------------------------------------------------------

extern "C" void kernel_launch(void* const* d_in, const int* in_sizes, int n_in,
                              void* d_out, int out_size, void* d_ws, size_t ws_size,
                              hipStream_t stream) {
  (void)in_sizes; (void)n_in; (void)out_size; (void)ws_size;
  const float* hidden = (const float*)d_in[0];
  const float* rope   = (const float*)d_in[1];
  const float* w_pre  = (const float*)d_in[2];
  const float* w_qkv  = (const float*)d_in[3];
  const float* w_qn   = (const float*)d_in[4];
  const float* w_kn   = (const float*)d_in[5];
  const float* w_proj = (const float*)d_in[6];
  const int*   invp   = (const int*)d_in[8];
  float* out = (float*)d_out;

  char* p = (char*)d_ws;
  f16* hbf    = (f16*)p;   p += (size_t)N_TOK * HID * sizeof(f16);
  f16* wqkvh  = (f16*)p;   p += (size_t)2 * QKV_OUT * HID * sizeof(f16);
  f16* wprojh = (f16*)p;   p += (size_t)2 * HID * HID * sizeof(f16);
  float* qkvf = (float*)p; p += (size_t)N_TOK * QKV_OUT * sizeof(float);
  f16* qh     = (f16*)p;   p += (size_t)HQ * N_TOK * HD * sizeof(f16);
  f16* kh     = (f16*)p;   p += (size_t)HKV * N_TOK * HD * sizeof(f16);
  f16* vh     = (f16*)p;   p += (size_t)HKV * N_TOK * HD * sizeof(f16);
  f16* obf    = (f16*)p;   p += (size_t)N_TOK * HQ * HD * sizeof(f16);

  rmsnorm_pre<<<N_TOK, 256, 0, stream>>>(hidden, w_pre, hbf);
  cast_f32_f16<<<2048, 256, 0, stream>>>(w_qkv, wqkvh, 2 * QKV_OUT * HID);
  cast_f32_f16<<<2048, 256, 0, stream>>>(w_proj, wprojh, 2 * HID * HID);

  dim3 gq((QKV_OUT + GEMM_BN - 1) / GEMM_BN, S0 / GEMM_BM);
  gemm_f16_wmma<<<gq, 256, 0, stream>>>(hbf, wqkvh, qkvf, S0, QKV_OUT, HID);
  gemm_f16_wmma<<<gq, 256, 0, stream>>>(hbf + (size_t)S0 * HID,
                                        wqkvh + (size_t)QKV_OUT * HID,
                                        qkvf + (size_t)S0 * QKV_OUT, S0, QKV_OUT, HID);

  qkv_post<<<N_TOK, 256, 0, stream>>>(qkvf, rope, w_qn, w_kn, invp, qh, kh, vh);

  flash_attn<<<dim3(N_TOK / 32, HKV), 256, 0, stream>>>(qh, kh, vh, qkvf, invp, obf);

  dim3 gp((HID + GEMM_BN - 1) / GEMM_BN, S0 / GEMM_BM);
  gemm_f16_wmma<<<gp, 256, 0, stream>>>(obf, wprojh, out, S0, HID, HID);
  gemm_f16_wmma<<<gp, 256, 0, stream>>>(obf + (size_t)S0 * HQ * HD,
                                        wprojh + (size_t)HID * HID,
                                        out + (size_t)S0 * HID, S0, HID, HID);
}
